// MultiHeadAttention_32177894981655
// MI455X (gfx1250) — matrix operand order repaired and checked by reading, not compiled, not run on hardware
//
#include <hip/hip_runtime.h>
#include <hip/hip_bf16.h>

#define B_  4
#define S_  2048
#define H_  8
#define D_  512
#define DK_ 64

typedef __attribute__((ext_vector_type(16))) _Float16 v16h;
typedef __attribute__((ext_vector_type(8)))  _Float16 v8h;
typedef __attribute__((ext_vector_type(8)))  float    v8f;
typedef _Float16 f16;

// ---------------------------------------------------------------------------
// WMMA helpers (CDNA5 wave32, V_WMMA_F32_16X16X32_F16)
// ---------------------------------------------------------------------------
__device__ __forceinline__ v8f wmma_f16(v16h a, v16h b, v8f c) {
  // 8 args: (neg_a, A, neg_b, B, c_mod, C, reuse_a, reuse_b)
  return __builtin_amdgcn_wmma_f32_16x16x32_f16(false, a, false, b, (short)0, c,
                                                false, false);
}

// A-matrix 16x32 f16 from row-major src (ld in elements).
// ISA layout: lane(0-15)=row M, hi half-wave shifts K by 8;
// VGPRs 0-3: K = hi*8 + 0..7, VGPRs 4-7: K = 16 + hi*8 + 0..7.
__device__ __forceinline__ v16h load_a16(const f16* base, int ld, int m0, int k0,
                                         int lane) {
  int m  = m0 + (lane & 15);
  int hi = (lane >> 4) & 1;
  const f16* p = base + (size_t)m * ld + k0 + hi * 8;
  v8h lo = *(const v8h*)p;
  v8h up = *(const v8h*)(p + 16);
  v16h a;
#pragma unroll
  for (int i = 0; i < 8; ++i) { a[i] = lo[i]; a[8 + i] = up[i]; }
  return a;
}

// B-matrix 32x16 f16 from "BT" storage: row-major [n][k] (ld in elements).
// ISA layout: lane(0-15)=col N, hi half-wave shifts K by 8;
// VGPRs 0-3: K = hi*8 + 0..7, VGPRs 4-7: K = 16 + hi*8 + 0..7.
__device__ __forceinline__ v16h load_bt16(const f16* bt, int ld, int n0, int k0,
                                          int lane) {
  int n  = n0 + (lane & 15);
  int hi = (lane >> 4) & 1;
  const f16* p = bt + (size_t)n * ld + k0 + hi * 8;
  v8h lo = *(const v8h*)p;
  v8h up = *(const v8h*)(p + 16);
  v16h b;
#pragma unroll
  for (int i = 0; i < 8; ++i) { b[i] = lo[i]; b[8 + i] = up[i]; }
  return b;
}

// ---------------------------------------------------------------------------
// Prep kernels
// ---------------------------------------------------------------------------
__global__ void cvt_f32_to_f16(const float* __restrict__ src, f16* __restrict__ dst,
                               int n) {
  int i = blockIdx.x * blockDim.x + threadIdx.x;
  int stride = gridDim.x * blockDim.x;
  for (; i < n; i += stride) dst[i] = (f16)src[i];
}

// src [Hn][Dn][En] fp32 -> dst [Hn*En][Dn] f16 (transposed per head)
__global__ void transpose_w(const float* __restrict__ src, f16* __restrict__ dst,
                            int Hn, int Dn, int En) {
  int i = blockIdx.x * blockDim.x + threadIdx.x;
  int stride = gridDim.x * blockDim.x;
  int total = Hn * Dn * En;
  for (; i < total; i += stride) {
    int e  = i % En;
    int d  = (i / En) % Dn;
    int hh = i / (En * Dn);
    dst[((size_t)hh * En + e) * Dn + d] = (f16)src[i];
  }
}

// ---------------------------------------------------------------------------
// GEMM: C[M x N] = A16[M x K] * BT16[N x K]^T + bias, wave = 16x64 tile
// ---------------------------------------------------------------------------
#define OUT_QK  0   // f16, [b][h][s][64] layout (m = b*2048+s, n = h*64+e)
#define OUT_VT  1   // f16, [b][h][64][S]  layout (transposed for P*V B-operand)
#define OUT_F32 2   // fp32 row-major [M][N]

__global__ void __launch_bounds__(128) gemm_wmma(
    const f16* __restrict__ A, int lda, const f16* __restrict__ BT, int ldb,
    const float* __restrict__ bias, void* __restrict__ out,
    int M, int N, int K, int mode) {
  int lane = threadIdx.x & 31;
  int wave = (blockIdx.x * blockDim.x + threadIdx.x) >> 5;
  int ntiles = N >> 6;
  int mt = wave / ntiles;
  int nt = wave - mt * ntiles;
  int m0 = mt << 4, n0 = nt << 6;
  if (m0 >= M) return;

  v8f acc[4] = {};
  for (int k0 = 0; k0 < K; k0 += 32) {
    v16h a = load_a16(A, lda, m0, k0, lane);
#pragma unroll
    for (int t = 0; t < 4; ++t) {
      v16h b = load_bt16(BT, ldb, n0 + t * 16, k0, lane);
      acc[t] = wmma_f16(a, b, acc[t]);
    }
  }

  int hi = (lane >> 4) & 1;
  int cl = lane & 15;
#pragma unroll
  for (int t = 0; t < 4; ++t) {
    int n = n0 + t * 16 + cl;
    float bb = bias[n];
    if (mode == OUT_VT) {
      int b_ = m0 >> 11;                 // m = b*2048 + s
      int s0 = (m0 & 2047) + hi * 8;     // 8 consecutive s per lane
      int h = n >> 6, e = n & 63;
      v8h pk;
#pragma unroll
      for (int r = 0; r < 8; ++r) pk[r] = (f16)(acc[t][r] + bb);
      *(v8h*)((f16*)out + (((size_t)(b_ * H_ + h) * DK_ + e) * S_ + s0)) = pk;
    } else if (mode == OUT_QK) {
      int b_ = m0 >> 11;
      int h = n >> 6, e = n & 63;
#pragma unroll
      for (int r = 0; r < 8; ++r) {
        int s = (m0 & 2047) + hi * 8 + r;
        ((f16*)out)[(((size_t)(b_ * H_ + h) * S_ + s) * DK_) + e] =
            (f16)(acc[t][r] + bb);
      }
    } else {
#pragma unroll
      for (int r = 0; r < 8; ++r) {
        int m = m0 + hi * 8 + r;
        ((float*)out)[(size_t)m * N + n] = acc[t][r] + bb;
      }
    }
  }
}

// ---------------------------------------------------------------------------
// Flash attention: one wave owns 16 query rows; key tiles of 32, online softmax
// ---------------------------------------------------------------------------
__global__ void __launch_bounds__(128) attn_wmma(
    const f16* __restrict__ Q, const f16* __restrict__ Kb,
    const f16* __restrict__ VT, f16* __restrict__ attn16,
    const int* __restrict__ has_mask) {
  __shared__ float pbuf_all[4][16 * 32];   // per-wave P transpose buffer
  int lane = threadIdx.x & 31;
  int wib  = threadIdx.x >> 5;
  float* pbuf = pbuf_all[wib];

  int wave = blockIdx.x * 4 + wib;         // B*H*(S/16) = 4096 waves total
  int qt = wave & (S_ / 16 - 1);
  int h  = (wave >> 7) & (H_ - 1);
  int b  = wave >> 10;
  int row0 = qt << 4;

  const f16* Qh = Q  + (size_t)(b * H_ + h) * S_ * DK_;
  const f16* Kh = Kb + (size_t)(b * H_ + h) * S_ * DK_;
  const f16* Vh = VT + (size_t)(b * H_ + h) * DK_ * S_;
  int msk = has_mask[0];

  // Q tile (16 rows x 64) held in registers for the whole key loop
  v16h aq0 = load_a16(Qh, DK_, row0, 0, lane);
  v16h aq1 = load_a16(Qh, DK_, row0, 32, lane);

  v8f o[4] = {};
  float mrow[8], lrow[8];
#pragma unroll
  for (int r = 0; r < 8; ++r) { mrow[r] = -3.0e38f; lrow[r] = 0.f; }

  int hi = (lane >> 4) & 1;
  int cl = lane & 15;
  int jend = msk ? (row0 + 16) : S_;       // causal: only tiles with j <= row
  for (int j0 = 0; j0 < jend; j0 += 32) {
    v8f s0 = {}, s1 = {};
    s0 = wmma_f16(aq0, load_bt16(Kh, DK_, j0,      0,  lane), s0);
    s0 = wmma_f16(aq1, load_bt16(Kh, DK_, j0,      32, lane), s0);
    s1 = wmma_f16(aq0, load_bt16(Kh, DK_, j0 + 16, 0,  lane), s1);
    s1 = wmma_f16(aq1, load_bt16(Kh, DK_, j0 + 16, 32, lane), s1);

#pragma unroll
    for (int r = 0; r < 8; ++r) {
      int row = row0 + hi * 8 + r;         // C layout: VGPR r <-> row r+8*hi
      float e0 = s0[r] * 0.125f;           // 1/sqrt(64)
      float e1 = s1[r] * 0.125f;
      if (msk) {
        e0 = (j0 + cl      > row) ? -3.0e38f : e0;
        e1 = (j0 + 16 + cl > row) ? -3.0e38f : e1;
      }
      float rmax = fmaxf(e0, e1);
#pragma unroll
      for (int d = 1; d < 16; d <<= 1)
        rmax = fmaxf(rmax, __shfl_xor(rmax, d, 32));
      float mnew = fmaxf(mrow[r], rmax);
      float corr = __expf(mrow[r] - mnew);
      mrow[r] = mnew;
      float p0 = __expf(e0 - mnew);
      float p1 = __expf(e1 - mnew);
      float rs = p0 + p1;
#pragma unroll
      for (int d = 1; d < 16; d <<= 1) rs += __shfl_xor(rs, d, 32);
      lrow[r] = lrow[r] * corr + rs;
      o[0][r] *= corr; o[1][r] *= corr; o[2][r] *= corr; o[3][r] *= corr;
      int prow = hi * 8 + r;
      pbuf[prow * 32 + cl]      = p0;      // stage P for A-layout re-read
      pbuf[prow * 32 + 16 + cl] = p1;
    }
    asm volatile("s_wait_dscnt 0" ::: "memory");  // LDS store->load, same wave

    // Re-read P as a 16x32 A operand
    v16h ap;
    {
      const float* pr = &pbuf[(size_t)cl * 32 + hi * 8];
#pragma unroll
      for (int i = 0; i < 8; ++i) ap[i] = (f16)pr[i];
      pr += 16;
#pragma unroll
      for (int i = 0; i < 8; ++i) ap[8 + i] = (f16)pr[i];
    }
#pragma unroll
    for (int t = 0; t < 4; ++t)
      o[t] = wmma_f16(ap, load_bt16(Vh, S_, t * 16, j0, lane), o[t]);
  }

#pragma unroll
  for (int r = 0; r < 8; ++r) lrow[r] = 1.0f / lrow[r];
#pragma unroll
  for (int t = 0; t < 4; ++t) {
    int col = h * DK_ + t * 16 + cl;       // concat heads: [b][s][h*64+e]
#pragma unroll
    for (int r = 0; r < 8; ++r) {
      int s = row0 + hi * 8 + r;
      attn16[((size_t)(b * S_ + s)) * (H_ * DK_) + col] = (f16)(o[t][r] * lrow[r]);
    }
  }
}

// ---------------------------------------------------------------------------
// Launch
// ---------------------------------------------------------------------------
extern "C" void kernel_launch(void* const* d_in, const int* in_sizes, int n_in,
                              void* d_out, int out_size, void* d_ws, size_t ws_size,
                              hipStream_t stream) {
  (void)in_sizes; (void)n_in; (void)out_size; (void)ws_size;
  const float* queries = (const float*)d_in[0];
  const float* keys    = (const float*)d_in[1];
  const float* values  = (const float*)d_in[2];
  const float* wq = (const float*)d_in[3];
  const float* bq = (const float*)d_in[4];
  const float* wk = (const float*)d_in[5];
  const float* bk = (const float*)d_in[6];
  const float* wv = (const float*)d_in[7];
  const float* bv = (const float*)d_in[8];
  const float* wo = (const float*)d_in[9];
  const float* bo = (const float*)d_in[10];
  const int* has_mask = (const int*)d_in[11];

  char* ws = (char*)d_ws;
  size_t off = 0;
  auto carve = [&](size_t bytes) {
    char* p = ws + off;
    off += (bytes + 255) & ~(size_t)255;
    return p;
  };
  const size_t act = (size_t)B_ * S_ * D_;           // 4,194,304 elems
  f16* q16  = (f16*)carve(act * 2);
  f16* k16  = (f16*)carve(act * 2);
  f16* v16_ = (f16*)carve(act * 2);
  f16* wqT  = (f16*)carve((size_t)D_ * D_ * 2);
  f16* wkT  = (f16*)carve((size_t)D_ * D_ * 2);
  f16* wvT  = (f16*)carve((size_t)D_ * D_ * 2);
  f16* woT  = (f16*)carve((size_t)D_ * D_ * 2);
  f16* Qb   = (f16*)carve(act * 2);                  // [b][h][s][64]
  f16* Kbf  = (f16*)carve(act * 2);                  // [b][h][s][64]
  f16* VTb  = (f16*)carve(act * 2);                  // [b][h][64][S]
  f16* at16 = (f16*)carve(act * 2);                  // [b][s][512]

  int nact = (int)act;
  cvt_f32_to_f16<<<2048, 256, 0, stream>>>(queries, q16, nact);
  cvt_f32_to_f16<<<2048, 256, 0, stream>>>(keys,    k16, nact);
  cvt_f32_to_f16<<<2048, 256, 0, stream>>>(values,  v16_, nact);
  transpose_w<<<512, 256, 0, stream>>>(wq, wqT, H_, D_, DK_);
  transpose_w<<<512, 256, 0, stream>>>(wk, wkT, H_, D_, DK_);
  transpose_w<<<512, 256, 0, stream>>>(wv, wvT, H_, D_, DK_);
  transpose_w<<<512, 256, 0, stream>>>(wo, woT, 1,  D_, D_);

  // (8192/16) m-tiles * (512/64) n-tiles = 4096 waves = 1024 blocks of 4 waves
  int gblocks = ((B_ * S_ / 16) * (D_ / 64)) / 4;
  gemm_wmma<<<gblocks, 128, 0, stream>>>(q16,  D_, wqT, D_, bq, Qb,   B_ * S_, D_, D_, OUT_QK);
  gemm_wmma<<<gblocks, 128, 0, stream>>>(k16,  D_, wkT, D_, bk, Kbf,  B_ * S_, D_, D_, OUT_QK);
  gemm_wmma<<<gblocks, 128, 0, stream>>>(v16_, D_, wvT, D_, bv, VTb,  B_ * S_, D_, D_, OUT_VT);

  attn_wmma<<<1024, 128, 0, stream>>>(Qb, Kbf, VTb, at16, has_mask);

  gemm_wmma<<<gblocks, 128, 0, stream>>>(at16, D_, woT, D_, bo, d_out, B_ * S_, D_, D_, OUT_F32);
}
